// ChildSumTreeLSTM_10179072491854
// MI455X (gfx1250) — matrix operand order repaired and checked by reading, not compiled, not run on hardware
//
#include <hip/hip_runtime.h>
#include <hip/hip_bf16.h>
#include <math.h>

#define N_NODES 8191
#define IN_DIM  1024
#define MEM_DIM 512
#define XCOLS   2048   // columns 0..1535 = iou pre-acts, 1536..2047 = xf

typedef __attribute__((ext_vector_type(16))) __bf16 bf16x16;
typedef __attribute__((ext_vector_type(8)))  __bf16 bf16x8;
typedef __attribute__((ext_vector_type(8)))  float  v8f;

__device__ __forceinline__ unsigned short f32_to_bf16_bits(float f) {
  unsigned int u = __float_as_uint(f);
  unsigned int r = u + 0x7fffu + ((u >> 16) & 1u);  // round-to-nearest-even
  return (unsigned short)(r >> 16);
}

__device__ __forceinline__ float sigmoidf_(float x) {
  return 1.0f / (1.0f + __expf(-x));
}

// Fragment loads (layouts per ISA 7.12.2, wave32):
//  A 16x32 bf16: lane -> row m = lane&15, half = lane>>4;
//     slots e<8 : k = k0 + half*8 + e ; slots e>=8 : k = k0 + 16 + half*8 + (e-8)
//  B 32x16 bf16: lane -> col n = lane&15; same k slots as A (e<8: k0+half*8+e; e>=8: k0+16+half*8+(e-8))
// Base pointers below already include the half-dependent offset.
__device__ __forceinline__ bf16x16 load_a_frag(const unsigned short* base, int k0) {
  bf16x8 a0 = *(const bf16x8*)(base + k0);
  bf16x8 a1 = *(const bf16x8*)(base + k0 + 16);
  return __builtin_shufflevector(a0, a1, 0,1,2,3,4,5,6,7,8,9,10,11,12,13,14,15);
}
__device__ __forceinline__ bf16x16 load_b_frag(const unsigned short* base, int k0) {
  bf16x8 b0 = *(const bf16x8*)(base + k0);
  bf16x8 b1 = *(const bf16x8*)(base + k0 + 16);
  return __builtin_shufflevector(b0, b1, 0,1,2,3,4,5,6,7,8,9,10,11,12,13,14,15);
}

__device__ __forceinline__ void load_set(bf16x16 a[2], bf16x16 b[4],
                                         const unsigned short* const Ab[2],
                                         const unsigned short* const Bb[4],
                                         int k0) {
#pragma unroll
  for (int i = 0; i < 2; ++i) a[i] = load_a_frag(Ab[i], k0);
#pragma unroll
  for (int j = 0; j < 4; ++j) b[j] = load_b_frag(Bb[j], k0);
}

__device__ __forceinline__ void wmma_set(v8f acc[2][4],
                                         const bf16x16 a[2], const bf16x16 b[4]) {
#pragma unroll
  for (int i = 0; i < 2; ++i)
#pragma unroll
    for (int j = 0; j < 4; ++j)
      acc[i][j] = __builtin_amdgcn_wmma_f32_16x16x32_bf16(
          /*neg_a=*/false, a[i], /*neg_b=*/false, b[j],
          /*c_mod=*/(short)0, acc[i][j], /*reuse_a=*/false, /*reuse_b=*/false);
}

// ---------------------------------------------------------------- conversions
__global__ void cvt_f32_bf16(const float* __restrict__ src,
                             unsigned short* __restrict__ dst, int n) {
  int i = blockIdx.x * blockDim.x + threadIdx.x;
  if (i < n) dst[i] = f32_to_bf16_bits(src[i]);
}

__global__ void concat_bias(const float* __restrict__ b_iou,
                            const float* __restrict__ b_f,
                            float* __restrict__ out) {
  int i = blockIdx.x * blockDim.x + threadIdx.x;
  if (i < 3 * MEM_DIM)      out[i] = b_iou[i];
  else if (i < XCOLS)       out[i] = b_f[i - 3 * MEM_DIM];
}

// ---------------------------------------------------------------- WMMA GEMM
// C[M,N] (f32) = A[M,K](bf16,row-major) * B[N,K](bf16,row-major)^T (+bias | +=C)
// One wave computes a 32x64 tile of C: 2 M-tiles x 4 N-tiles = 8 independent
// accumulators, software-pipelined K loop (double-buffered fragments) so the
// next stage's global loads overlap the current stage's 8 v_wmma ops.
// Requires: N % 64 == 0, K % 64 == 0 (true here: N in {2048,1536,512},
// K in {1024,512}).
__global__ void __launch_bounds__(256)
wmma_gemm_bf16(const unsigned short* __restrict__ A,
               const unsigned short* __restrict__ B,
               float* __restrict__ C,
               const float* __restrict__ bias,
               int M, int N, int K, int lda, int ldb, int ldc, int accumulate)
{
  const int wave = threadIdx.x >> 5;
  const int lane = threadIdx.x & 31;
  const int qn = N >> 6;                    // 64-wide column tiles
  const int tiles_m = (M + 31) >> 5;        // 32-tall row tiles
  const int tile = blockIdx.x * 8 + wave;
  if (tile >= tiles_m * qn) return;         // wave-uniform exit
  const int tm = tile / qn;
  const int tq = tile - tm * qn;

  const int ln15 = lane & 15;
  const int half = lane >> 4;

  const unsigned short* Ab[2];
#pragma unroll
  for (int i = 0; i < 2; ++i) {
    int ar = tm * 32 + i * 16 + ln15;
    if (ar >= M) ar = M - 1;                // clamp: rows >= M never stored
    Ab[i] = A + (size_t)ar * lda + half * 8;
  }

  int ncol[4];
  const unsigned short* Bb[4];
#pragma unroll
  for (int j = 0; j < 4; ++j) {
    ncol[j] = (tq * 4 + j) * 16 + ln15;
    Bb[j] = B + (size_t)ncol[j] * ldb + half * 8;
  }

  v8f acc[2][4];
  if (accumulate) {
#pragma unroll
    for (int i = 0; i < 2; ++i)
#pragma unroll
      for (int j = 0; j < 4; ++j)
#pragma unroll
        for (int r = 0; r < 8; ++r) {
          int m = tm * 32 + i * 16 + half * 8 + r;
          acc[i][j][r] = (m < M) ? C[(size_t)m * ldc + ncol[j]] : 0.0f;
        }
  } else {
#pragma unroll
    for (int i = 0; i < 2; ++i)
#pragma unroll
      for (int j = 0; j < 4; ++j) {
        float b0 = bias ? bias[ncol[j]] : 0.0f;
#pragma unroll
        for (int r = 0; r < 8; ++r) acc[i][j][r] = b0;
      }
  }

  // Software-pipelined K loop: two live fragment sets.
  bf16x16 aX[2], bX[4], aY[2], bY[4];
  load_set(aX, bX, Ab, Bb, 0);
  for (int k0 = 0; k0 < K; k0 += 64) {
    load_set(aY, bY, Ab, Bb, k0 + 32);
    wmma_set(acc, aX, bX);
    int kn = (k0 + 64 < K) ? (k0 + 64) : 0;   // last prefetch is a harmless dummy
    load_set(aX, bX, Ab, Bb, kn);
    wmma_set(acc, aY, bY);
  }

#pragma unroll
  for (int i = 0; i < 2; ++i)
#pragma unroll
    for (int j = 0; j < 4; ++j)
#pragma unroll
      for (int r = 0; r < 8; ++r) {
        int m = tm * 32 + i * 16 + half * 8 + r;
        if (m < M) C[(size_t)m * ldc + ncol[j]] = acc[i][j][r];
      }
}

// ------------------------------------------------- gather masked child states
__global__ void gather_children(const float* __restrict__ h_all,
                                const int* __restrict__ cidx,
                                const float* __restrict__ cmask,
                                int t0, int L,
                                unsigned short* __restrict__ CH,   // [2L,512] bf16
                                unsigned short* __restrict__ HS)   // [L,512]  bf16
{
  int gid = blockIdx.x * blockDim.x + threadIdx.x;
  if (gid >= L * MEM_DIM) return;
  int i = gid >> 9;
  int m = gid & (MEM_DIM - 1);
  int t = t0 + i;
  int   c0 = cidx[2 * t + 0], c1 = cidx[2 * t + 1];
  float m0 = cmask[2 * t + 0], m1 = cmask[2 * t + 1];
  float h0 = h_all[(size_t)c0 * MEM_DIM + m] * m0;
  float h1 = h_all[(size_t)c1 * MEM_DIM + m] * m1;
  CH[(size_t)(2 * i + 0) * MEM_DIM + m] = f32_to_bf16_bits(h0);
  CH[(size_t)(2 * i + 1) * MEM_DIM + m] = f32_to_bf16_bits(h1);
  HS[(size_t)i * MEM_DIM + m] = f32_to_bf16_bits(h0 + h1);
}

// --------------------------------------------------------- gate/state update
__global__ void node_update(const float* __restrict__ XB,
                            const float* __restrict__ FPRE,   // [2L,512] raw ch@Wfh^T
                            const int* __restrict__ cidx,
                            const float* __restrict__ cmask,
                            int t0, int L, int has_children,
                            float* __restrict__ c_all, float* __restrict__ h_all)
{
  int gid = blockIdx.x * blockDim.x + threadIdx.x;
  if (gid >= L * MEM_DIM) return;
  int i = gid >> 9;
  int m = gid & (MEM_DIM - 1);
  int t = t0 + i;
  const float* xrow = XB + (size_t)t * XCOLS;
  float iv = sigmoidf_(xrow[m]);
  float ov = sigmoidf_(xrow[MEM_DIM + m]);
  float uv = tanhf(xrow[2 * MEM_DIM + m]);
  float fc = 0.0f;
  if (has_children) {
    float xf = xrow[3 * MEM_DIM + m];
#pragma unroll
    for (int s = 0; s < 2; ++s) {
      int   cs = cidx[2 * t + s];
      float ms = cmask[2 * t + s];
      float f  = sigmoidf_(FPRE[(size_t)(2 * i + s) * MEM_DIM + m] + xf);
      fc += f * c_all[(size_t)cs * MEM_DIM + m] * ms;
    }
  }
  float c = iv * uv + fc;
  c_all[(size_t)t * MEM_DIM + m] = c;
  h_all[(size_t)t * MEM_DIM + m] = ov * tanhf(c);
}

// ------------------------------------------------------------------ launcher
extern "C" void kernel_launch(void* const* d_in, const int* in_sizes, int n_in,
                              void* d_out, int out_size, void* d_ws, size_t ws_size,
                              hipStream_t stream)
{
  (void)in_sizes; (void)n_in; (void)out_size; (void)ws_size;
  const float* inputs = (const float*)d_in[0];
  const float* W_ioux = (const float*)d_in[1];
  const float* b_ioux = (const float*)d_in[2];
  const float* W_iouh = (const float*)d_in[3];
  const float* W_fx   = (const float*)d_in[4];
  const float* b_fx   = (const float*)d_in[5];
  const float* W_fh   = (const float*)d_in[6];
  const int*   cidx   = (const int*)d_in[7];
  const float* cmask  = (const float*)d_in[8];

  float* c_all = (float*)d_out;                       // [8191,512]
  float* h_all = c_all + (size_t)N_NODES * MEM_DIM;   // [8191,512]

  char* ws = (char*)d_ws;
  size_t off = 0;
  auto wsalloc = [&](size_t bytes) -> void* {
    void* p = (void*)(ws + off);
    off = (off + bytes + 255) & ~(size_t)255;
    return p;
  };
  unsigned short* in_bf  = (unsigned short*)wsalloc((size_t)N_NODES * IN_DIM * 2);
  unsigned short* Wx_bf  = (unsigned short*)wsalloc((size_t)XCOLS * IN_DIM * 2);
  unsigned short* Wh_bf  = (unsigned short*)wsalloc((size_t)3 * MEM_DIM * MEM_DIM * 2);
  unsigned short* Wfh_bf = (unsigned short*)wsalloc((size_t)MEM_DIM * MEM_DIM * 2);
  float*          bias   = (float*)wsalloc((size_t)XCOLS * 4);
  float*          XB     = (float*)wsalloc((size_t)N_NODES * XCOLS * 4);
  unsigned short* HS     = (unsigned short*)wsalloc((size_t)2048 * MEM_DIM * 2);
  unsigned short* CH     = (unsigned short*)wsalloc((size_t)4096 * MEM_DIM * 2);
  float*          FPRE   = (float*)wsalloc((size_t)4096 * MEM_DIM * 4);

  auto cvt = [&](const float* s, unsigned short* d, int n) {
    cvt_f32_bf16<<<(n + 255) / 256, 256, 0, stream>>>(s, d, n);
  };
  cvt(inputs, in_bf, N_NODES * IN_DIM);
  cvt(W_ioux, Wx_bf, 3 * MEM_DIM * IN_DIM);
  cvt(W_fx,   Wx_bf + (size_t)3 * MEM_DIM * IN_DIM, MEM_DIM * IN_DIM);
  cvt(W_iouh, Wh_bf, 3 * MEM_DIM * MEM_DIM);
  cvt(W_fh,   Wfh_bf, MEM_DIM * MEM_DIM);
  concat_bias<<<(XCOLS + 255) / 256, 256, 0, stream>>>(b_ioux, b_fx, bias);

  auto gemm = [&](const unsigned short* A, const unsigned short* B, float* C,
                  const float* bvec, int M, int N, int K,
                  int lda, int ldb, int ldc, int accum) {
    int tiles = ((M + 31) >> 5) * (N >> 6);   // 32x64 tile per wave
    int blocks = (tiles + 7) / 8;
    wmma_gemm_bf16<<<blocks, 256, 0, stream>>>(A, B, C, bvec, M, N, K,
                                               lda, ldb, ldc, accum);
  };

  // Phase 1: XB[t, 0:2048] = inputs @ [W_ioux;W_fx]^T + [b_ioux;b_fx]
  gemm(in_bf, Wx_bf, XB, bias, N_NODES, XCOLS, IN_DIM, IN_DIM, IN_DIM, XCOLS, 0);

  // Phase 2: leaves (t = 0..4095): no children, iou = xiou, c = i*u.
  {
    int L = 4096, t0 = 0, n = L * MEM_DIM;
    node_update<<<(n + 255) / 256, 256, 0, stream>>>(
        XB, FPRE, cidx, cmask, t0, L, 0, c_all, h_all);
  }

  // Phase 3: 12 internal levels, bottom-up. Level k: L = 2^(12-k) nodes,
  // contiguous scan range starting at t0 = 8192 - 2^(13-k).
  for (int k = 1; k <= 12; ++k) {
    int L  = 1 << (12 - k);
    int t0 = 8192 - (1 << (13 - k));
    int n  = L * MEM_DIM;
    gather_children<<<(n + 255) / 256, 256, 0, stream>>>(
        h_all, cidx, cmask, t0, L, CH, HS);
    // XB[t0: t0+L, 0:1536] += HS @ W_iouh^T   (accumulate in place, ldc=2048)
    gemm(HS, Wh_bf, XB + (size_t)t0 * XCOLS, nullptr,
         L, 3 * MEM_DIM, MEM_DIM, MEM_DIM, MEM_DIM, XCOLS, 1);
    // FPRE[0:2L, :] = CH @ W_fh^T
    gemm(CH, Wfh_bf, FPRE, nullptr,
         2 * L, MEM_DIM, MEM_DIM, MEM_DIM, MEM_DIM, MEM_DIM, 0);
    node_update<<<(n + 255) / 256, 256, 0, stream>>>(
        XB, FPRE, cidx, cmask, t0, L, 1, c_all, h_all);
  }
}
